// GuidedSampler_30399778521730
// MI455X (gfx1250) — matrix operand order repaired and checked by reading, not compiled, not run on hardware
//
#include <hip/hip_runtime.h>

// ---------------------------------------------------------------------------
// GuidedSampler for MI455X (gfx1250, wave32, WMMA)
//   features [32,768,32,32] f32, query [32,16,32,32] f32, W [64,16,768] f32
//   out: sel [32,16,32,32] f32 | codes [32] (as f32) | commit_loss [1]
// ---------------------------------------------------------------------------

typedef __attribute__((ext_vector_type(16))) __bf16 v16bf;
typedef __attribute__((ext_vector_type(2)))  __bf16 v2bf;
typedef __attribute__((ext_vector_type(8)))  float  v8f;

#define B_   32
#define C_   768
#define HW_  1024
#define K_   64
#define DQ_  16
#define CCH  24          // C chunks of 32
#define HWH  512         // hw half width (two halves of 1024)
#define NCHUNK (2 * CCH) // linearized (half, cc) chunk count

union FragU { v16bf v; uint4 q[2]; };

// Pack two f32 -> bf16x2 dword.  Single v_perm_b32 (truncation) unless the
// packed RNE convert builtin exists on this toolchain.
__device__ __forceinline__ unsigned pack_bf16x2(float lo, float hi) {
#if __has_builtin(__builtin_amdgcn_cvt_pk_bf16_f32)
  union { v2bf v; unsigned u; } c;
  c.v = __builtin_amdgcn_cvt_pk_bf16_f32(lo, hi);
  return c.u;
#else
  // bytes [hi.3, hi.2, lo.3, lo.2]
  return __builtin_amdgcn_perm(__float_as_uint(hi), __float_as_uint(lo),
                               0x07060302u);
#endif
}

// ---- register-staged chunk copy: F[b][c0..c0+31][hwbase..+511] -> bf16 LDS
// Unit = (c-pair, 4 hw): two row-wise float4 loads (coalesced), 4 packed
// converts, 4 dword LDS stores.  8 units per thread (256 threads, 2048 units).
struct StageRegs { float4 r0[8]; float4 r1[8]; };

__device__ __forceinline__ void stage_load(StageRegs& s,
                                           const float* __restrict__ src,
                                           int tid) {
#pragma unroll
  for (int u = 0; u < 8; ++u) {
    const int i  = tid + u * 256;
    const int cp = i >> 7;
    const int hw4 = i & 127;
    s.r0[u] = ((const float4*)(src + (size_t)(2 * cp)     * HW_))[hw4];
    s.r1[u] = ((const float4*)(src + (size_t)(2 * cp + 1) * HW_))[hw4];
  }
}

__device__ __forceinline__ void stage_store(const StageRegs& s,
                                            unsigned short* ldsB, int tid) {
  unsigned* ldsW = (unsigned*)ldsB;
#pragma unroll
  for (int u = 0; u < 8; ++u) {
    const int i  = tid + u * 256;
    const int cp = i >> 7;
    const int hw4 = i & 127;
    const int base = (hw4 * 4) * 16 + cp;   // u32 units; LDS row = 16 dwords
    ldsW[base     ] = pack_bf16x2(s.r0[u].x, s.r1[u].x);
    ldsW[base + 16] = pack_bf16x2(s.r0[u].y, s.r1[u].y);
    ldsW[base + 32] = pack_bf16x2(s.r0[u].z, s.r1[u].z);
    ldsW[base + 48] = pack_bf16x2(s.r0[u].w, s.r1[u].w);
  }
}

// Non-pipelined variant (kernel 3)
__device__ __forceinline__ void stage_chunk(unsigned short* ldsB,
                                            const float* __restrict__ src,
                                            int tid) {
  StageRegs s;
  stage_load(s, src, tid);
  stage_store(s, ldsB, tid);
}

// A fragment (16x32 bf16): lane holds M=laneLo; K = e + 16*laneHi.
// Row-major A panel in LDS.
__device__ __forceinline__ v16bf load_a_frag(const unsigned short* ldsArow,
                                             int cc, int laneHi) {
  FragU a;
  const unsigned short* p = ldsArow + cc * 32 + laneHi * 16;
  a.q[0] = *(const uint4*)(p);
  a.q[1] = *(const uint4*)(p + 8);
  return a.v;
}

// B fragment (32x16 bf16): lane holds N=laneLo, K = e + 16*laneHi.
__device__ __forceinline__ v16bf load_b_frag(const unsigned short* ldsB,
                                             int n0, int laneLo, int laneHi) {
  FragU f;
  const unsigned short* p = ldsB + (n0 + laneLo) * 32 + laneHi * 16;
  f.q[0] = ((const uint4*)p)[0];
  f.q[1] = ((const uint4*)p)[1];
  return f.v;
}

// ---------------------------------------------------------------------------
// Kernel 1: one workgroup per (b, k-pair).  Waves 0-3 -> k0, waves 4-7 -> k1;
// each wave owns 128 columns (8 N-tiles) per hw-half.  Double-buffered LDS:
// chunk g+1 is loaded to registers and converted while chunk g feeds WMMAs.
// Fused (query-kv)^2 reduction, kv never materialized.
// ---------------------------------------------------------------------------
__global__ __launch_bounds__(256)
void kv_dist_kernel(const float* __restrict__ F, const float* __restrict__ Q,
                    const float* __restrict__ Wt, float* __restrict__ dist2) {
  __shared__ __align__(16) unsigned short ldsA[2 * DQ_ * C_];   // 48KB
  __shared__ __align__(16) unsigned short ldsB[2 * HWH * 32];   // 2x32KB ping-pong
  __shared__ float red[8];

  const int b      = blockIdx.x >> 5;
  const int k0     = (blockIdx.x & 31) << 1;
  const int tid    = threadIdx.x;
  const int wave   = tid >> 5;
  const int lane   = tid & 31;
  const int laneLo = lane & 15;
  const int laneHi = lane >> 4;
  const int kg     = wave >> 2;   // which k of the pair
  const int wl     = wave & 3;    // wave within k: 128-column slab

  const float* Fb  = F  + (size_t)b  * C_ * HW_;
  const float* Wk0 = Wt + (size_t)k0 * DQ_ * C_;   // two consecutive 16x768 panels
  const float* Qb  = Q  + (size_t)b  * DQ_ * HW_;

  // Stage both A panels (adjacent cols -> adjacent halves: packed stores)
  {
    const float2* wsrc = (const float2*)Wk0;
    unsigned* ldsAW = (unsigned*)ldsA;
    for (int i = tid; i < DQ_ * C_; i += 256) {    // 12288 dwords = 24576 bf16
      const float2 w2 = wsrc[i];
      ldsAW[i] = pack_bf16x2(w2.x, w2.y);
    }
  }

  const unsigned short* ldsArow = ldsA + (size_t)(kg * DQ_ + laneLo) * C_;
  const v8f vzero = {};
  v8f D[8] = {};
  float dacc = 0.f;
  StageRegs sreg;

  // Prologue: stage chunk 0 into buffer 0
  stage_load(sreg, Fb, tid);
  stage_store(sreg, ldsB, tid);

  for (int g = 0; g < NCHUNK; ++g) {
    __syncthreads();   // buffer g&1 staged; buffer (g+1)&1 free to write
    const int cc   = (g < CCH) ? g : g - CCH;
    const int half = (g < CCH) ? 0 : 1;

    // Issue global loads for chunk g+1 early (latency hidden behind WMMAs)
    if (g + 1 < NCHUNK) {
      const int gn  = g + 1;
      const int ccn = (gn < CCH) ? gn : gn - CCH;
      const int hbn = (gn < CCH) ? 0 : HWH;
      stage_load(sreg, Fb + (size_t)(ccn * 32) * HW_ + hbn, tid);
    }

    // 8 WMMAs on current buffer; B fragments pipelined one tile ahead
    const unsigned short* buf = ldsB + (size_t)(g & 1) * (HWH * 32);
    const v16bf a = load_a_frag(ldsArow, cc, laneHi);
    v16bf bcur = load_b_frag(buf, wl * 128, laneLo, laneHi);
#pragma unroll
    for (int nt = 0; nt < 8; ++nt) {
      v16bf bnext;
      if (nt < 7) bnext = load_b_frag(buf, wl * 128 + (nt + 1) * 16, laneLo, laneHi);
      D[nt] = __builtin_amdgcn_wmma_f32_16x16x32_bf16(
          false, a, false, bcur, (short)0, D[nt], false, false);
      if (nt < 7) bcur = bnext;
    }

    // End of a half: fused distance epilogue, reset accumulators
    if (cc == CCH - 1) {
      const int hwbase = half * HWH;
#pragma unroll
      for (int nt = 0; nt < 8; ++nt) {
        const int hw = hwbase + wl * 128 + nt * 16 + laneLo;
#pragma unroll
        for (int v = 0; v < 8; ++v) {
          const float qv = Qb[(size_t)(v + laneHi * 8) * HW_ + hw];
          const float d = qv - D[nt][v];
          dacc += d * d;
        }
        D[nt] = vzero;
      }
    }

    // Convert + store chunk g+1 into the other buffer
    if (g + 1 < NCHUNK)
      stage_store(sreg, ldsB + (size_t)((g + 1) & 1) * (HWH * 32), tid);
  }

  // Reduce: waves 0-3 -> dist2[b,k0], waves 4-7 -> dist2[b,k0+1]
  for (int off = 16; off > 0; off >>= 1) dacc += __shfl_down(dacc, off, 32);
  if (lane == 0) red[wave] = dacc;
  __syncthreads();
  if (tid == 0) {
    dist2[b * K_ + k0]     = red[0] + red[1] + red[2] + red[3];
    dist2[b * K_ + k0 + 1] = red[4] + red[5] + red[6] + red[7];
  }
}

// ---------------------------------------------------------------------------
// Kernel 2: argmin over k, codes + commit_loss.
// commit_loss = sum_b min_k dist2[b,k] / (B*DQ*HW)
// ---------------------------------------------------------------------------
__global__ __launch_bounds__(32)
void argmin_kernel(const float* __restrict__ dist2, int* __restrict__ codes_i,
                   float* __restrict__ out_codes, float* __restrict__ out_loss) {
  __shared__ float sm[B_];
  const int b = threadIdx.x;   // blockDim.x == 32
  const float* d = dist2 + b * K_;
  float best = 3.402823466e38f;
  int barg = 0;
  for (int k = 0; k < K_; ++k) {
    const float v = d[k];
    if (v < best) { best = v; barg = k; }
  }
  codes_i[b]   = barg;
  out_codes[b] = (float)barg;
  sm[b]        = best;
  __syncthreads();
  if (b == 0) {
    float s = 0.f;
    for (int i = 0; i < B_; ++i) s += sm[i];
    out_loss[0] = s / (float)(B_ * DQ_ * HW_);
  }
}

// ---------------------------------------------------------------------------
// Kernel 3: one workgroup per b — recompute kv only for k = codes[b] (1/64 of
// the GEMM work) and write sel[b,q,hw].  8 waves x 4 N-tiles per hw-half.
// ---------------------------------------------------------------------------
__global__ __launch_bounds__(256)
void kv_select_kernel(const float* __restrict__ F, const float* __restrict__ Wt,
                      const int* __restrict__ codes_i, float* __restrict__ sel) {
  __shared__ __align__(16) unsigned short ldsA[DQ_ * C_];   // 24KB
  __shared__ __align__(16) unsigned short ldsB[HWH * 32];   // 32KB

  const int b      = blockIdx.x;
  const int tid    = threadIdx.x;
  const int wave   = tid >> 5;
  const int lane   = tid & 31;
  const int laneLo = lane & 15;
  const int laneHi = lane >> 4;

  const int k = codes_i[b];
  const float* Fb = F  + (size_t)b * C_ * HW_;
  const float* Wk = Wt + (size_t)k * DQ_ * C_;

  {
    const float2* wsrc = (const float2*)Wk;
    unsigned* ldsAW = (unsigned*)ldsA;
    for (int i = tid; i < (DQ_ * C_) / 2; i += 256) {
      const float2 w2 = wsrc[i];
      ldsAW[i] = pack_bf16x2(w2.x, w2.y);
    }
  }

  const unsigned short* ldsArow = ldsA + (size_t)laneLo * C_;

  for (int half = 0; half < 2; ++half) {
    const int hwbase = half * HWH;
    v8f D[4] = {};   // 4 N-tiles per wave: 8 waves * 64 cols = 512

    for (int cc = 0; cc < CCH; ++cc) {
      __syncthreads();
      const float* src = Fb + (size_t)(cc * 32) * HW_ + hwbase;
      stage_chunk(ldsB, src, tid);
      if (cc + 1 < CCH)   // global_prefetch_b8 hint for next chunk
        __builtin_prefetch(src + 32 * HW_ + (tid << 4), 0, 0);
      __syncthreads();

      const v16bf a = load_a_frag(ldsArow, cc, laneHi);
#pragma unroll
      for (int nt = 0; nt < 4; ++nt) {
        const v16bf bf_ = load_b_frag(ldsB, wave * 64 + nt * 16, laneLo, laneHi);
        D[nt] = __builtin_amdgcn_wmma_f32_16x16x32_bf16(
            false, a, false, bf_, (short)0, D[nt], false, false);
      }
    }

    float* selb = sel + (size_t)b * DQ_ * HW_;
#pragma unroll
    for (int nt = 0; nt < 4; ++nt) {
      const int hw = hwbase + wave * 64 + nt * 16 + laneLo;
#pragma unroll
      for (int v = 0; v < 8; ++v) {
        selb[(size_t)(v + laneHi * 8) * HW_ + hw] = D[nt][v];
      }
    }
  }
}

// ---------------------------------------------------------------------------
extern "C" void kernel_launch(void* const* d_in, const int* in_sizes, int n_in,
                              void* d_out, int out_size, void* d_ws, size_t ws_size,
                              hipStream_t stream) {
  (void)in_sizes; (void)n_in; (void)out_size; (void)ws_size;
  const float* F  = (const float*)d_in[0];   // [32,768,32,32]
  const float* Q  = (const float*)d_in[1];   // [32,16,32,32]
  const float* Wt = (const float*)d_in[2];   // [64,16,768]
  float* out = (float*)d_out;                // sel | codes | loss

  float* dist2   = (float*)d_ws;                               // 2048 floats
  int*   codes_i = (int*)((char*)d_ws + 2048 * sizeof(float)); // 32 ints

  const int SEL_ELEMS = B_ * DQ_ * HW_;   // 524288

  kv_dist_kernel<<<(B_ * K_) / 2, 256, 0, stream>>>(F, Q, Wt, dist2);
  argmin_kernel<<<1, 32, 0, stream>>>(dist2, codes_i,
                                      out + SEL_ELEMS, out + SEL_ELEMS + B_);
  kv_select_kernel<<<B_, 256, 0, stream>>>(F, Wt, codes_i, out);
}
